// EMDLoss_55920474194485
// MI455X (gfx1250) — matrix operand order repaired and checked by reading, not compiled, not run on hardware
//
#include <hip/hip_runtime.h>
#include <hip/hip_bf16.h>

#define BSZ  128
#define NR   16
#define NW   48
#define DIM  1024
#define ITERS_ 50
#define NEG_INV_EPS (-20.0f)   // -1/0.05

typedef __attribute__((ext_vector_type(16))) _Float16 v16h;
typedef __attribute__((ext_vector_type(8)))  _Float16 v8h;
typedef __attribute__((ext_vector_type(8)))  float    v8f;

// ---------------------------------------------------------------------------
// One-time f32 -> f16 staging (each thread converts 8 contiguous elements).
// ---------------------------------------------------------------------------
__global__ __launch_bounds__(256) void cvt_f16_kernel(
    const float* __restrict__ src, _Float16* __restrict__ dst, int n8)
{
    const int idx = blockIdx.x * blockDim.x + threadIdx.x;
    if (idx >= n8) return;
    const float4* p = reinterpret_cast<const float4*>(src) + (size_t)idx * 2;
    const float4 f0 = p[0], f1 = p[1];
    v8h h;
    h[0] = (_Float16)f0.x; h[1] = (_Float16)f0.y;
    h[2] = (_Float16)f0.z; h[3] = (_Float16)f0.w;
    h[4] = (_Float16)f1.x; h[5] = (_Float16)f1.y;
    h[6] = (_Float16)f1.z; h[7] = (_Float16)f1.w;
    reinterpret_cast<v8h*>(dst)[idx] = h;
}

// ---------------------------------------------------------------------------
// Shared Sinkhorn (50 iters) + EMD reduction, runs on a 96-thread block with
// costL/Km already populated in LDS.
// ---------------------------------------------------------------------------
__device__ __forceinline__ void sinkhorn_and_emd(
    float (*costL)[NW], float (*Km)[NW],
    float* uL, float* vL, float* aw, float* bw, float* red,
    int tid, int kp, int sl, float* emd_out)
{
    if (tid < NR) { aw[tid] = (tid < kp) ? __fdividef(1.0f, (float)kp) : 0.0f; uL[tid] = 1.0f; }
    if (tid < NW) { bw[tid] = (tid < sl) ? __fdividef(1.0f, (float)sl) : 0.0f; vL[tid] = 1.0f; }
    __syncthreads();

    for (int it = 0; it < ITERS_; ++it) {
        if (tid < NR) {
            float kv = 0.f;
            #pragma unroll
            for (int n = 0; n < NW; ++n) kv += Km[tid][n] * vL[n];
            const float a_ = aw[tid];
            uL[tid] = (a_ > 0.f) ? __fdividef(a_, fmaxf(kv, 1e-12f)) : 0.f;
        }
        __syncthreads();
        if (tid < NW) {
            float ku = 0.f;
            #pragma unroll
            for (int m = 0; m < NR; ++m) ku += Km[m][tid] * uL[m];
            const float b_ = bw[tid];
            vL[tid] = (b_ > 0.f) ? __fdividef(b_, fmaxf(ku, 1e-12f)) : 0.f;
        }
        __syncthreads();
    }

    float part = 0.f;
    for (int t = tid; t < NR * NW; t += 96) {
        const int m = t / NW, n = t % NW;
        part += costL[m][n] * uL[m] * Km[m][n] * vL[n];
    }
    red[tid] = part;
    __syncthreads();
    if (tid == 0) {
        float sum = 0.f;
        #pragma unroll 8
        for (int t = 0; t < 96; ++t) sum += red[t];
        *emd_out = -sum;
    }
}

// ---------------------------------------------------------------------------
// Fast path: f16 inputs pre-staged in workspace.
// Inner loop per k-step: 4x global_load_b128 + 1x v_wmma_f32_16x16x32_f16.
// ---------------------------------------------------------------------------
__global__ __launch_bounds__(96) void emd_pair_f16_kernel(
    const _Float16* __restrict__ im_h, const _Float16* __restrict__ s_h,
    const int* __restrict__ s_l, const int* __restrict__ k_part,
    float* __restrict__ emd)
{
    const int i    = blockIdx.x;
    const int j    = blockIdx.y;
    const int tid  = threadIdx.x;
    const int wave = tid >> 5;
    const int lane = tid & 31;
    const int hl   = lane >> 4;
    const int l16  = lane & 15;

    __shared__ float costL[NR][NW];
    __shared__ float Km[NR][NW];
    __shared__ float uL[NR], vL[NW], aw[NR], bw[NW];
    __shared__ float red[96];

    const int wbase = wave * 16;
    const _Float16* __restrict__ arow = im_h + ((size_t)i * NR + l16) * DIM;
    const _Float16* __restrict__ brow = s_h  + ((size_t)j * NW + wbase + l16) * DIM;

    v8f acc = {0.f, 0.f, 0.f, 0.f, 0.f, 0.f, 0.f, 0.f};
    #pragma unroll 2
    for (int k0 = 0; k0 < DIM; k0 += 32) {
        const int ksA = k0 + hl * 8;    // A: lanes 0-15 -> K{+0..7,+16..23}; 16-31 -> K{+8..15,+24..31}
        const int ksB = k0 + hl * 8;    // B: lanes 0-15 -> K{+0..7,+16..23}; 16-31 -> K{+8..15,+24..31}
        const v8h a0 = *reinterpret_cast<const v8h*>(arow + ksA);
        const v8h a1 = *reinterpret_cast<const v8h*>(arow + ksA + 16);
        const v8h b0 = *reinterpret_cast<const v8h*>(brow + ksB);
        const v8h b1 = *reinterpret_cast<const v8h*>(brow + ksB + 16);
        v16h a, b;
        #pragma unroll
        for (int t = 0; t < 8; ++t) {
            a[t] = a0[t]; a[8 + t] = a1[t];
            b[t] = b0[t]; b[8 + t] = b1[t];
        }
        acc = __builtin_amdgcn_wmma_f32_16x16x32_f16(
                  false, a, false, b, (short)0, acc, false, false);
    }

    #pragma unroll
    for (int e = 0; e < 8; ++e) {
        const float d = acc[e];
        const float c = 1.0f - (d > 0.0f ? d : 0.0f);
        const int row = e + hl * 8;
        costL[row][wbase + l16] = c;
        Km[row][wbase + l16]    = __expf(c * NEG_INV_EPS);
    }

    sinkhorn_and_emd(costL, Km, uL, vL, aw, bw, red, tid,
                     k_part[i], s_l[j], &emd[i * BSZ + j]);
}

// ---------------------------------------------------------------------------
// Fallback path (ws too small): convert f32 -> f16 on the fly.
// ---------------------------------------------------------------------------
__device__ __forceinline__ void cvt8(v16h& d, int o, const float4 f0, const float4 f1) {
    d[o + 0] = (_Float16)f0.x; d[o + 1] = (_Float16)f0.y;
    d[o + 2] = (_Float16)f0.z; d[o + 3] = (_Float16)f0.w;
    d[o + 4] = (_Float16)f1.x; d[o + 5] = (_Float16)f1.y;
    d[o + 6] = (_Float16)f1.z; d[o + 7] = (_Float16)f1.w;
}

__global__ __launch_bounds__(96) void emd_pair_f32_kernel(
    const float* __restrict__ im, const float* __restrict__ s,
    const int* __restrict__ s_l, const int* __restrict__ k_part,
    float* __restrict__ emd)
{
    const int i    = blockIdx.x;
    const int j    = blockIdx.y;
    const int tid  = threadIdx.x;
    const int wave = tid >> 5;
    const int lane = tid & 31;
    const int hl   = lane >> 4;
    const int l16  = lane & 15;

    __shared__ float costL[NR][NW];
    __shared__ float Km[NR][NW];
    __shared__ float uL[NR], vL[NW], aw[NR], bw[NW];
    __shared__ float red[96];

    const int wbase = wave * 16;
    const float* __restrict__ arow = im + ((size_t)i * NR + l16) * DIM;
    const float* __restrict__ brow = s  + ((size_t)j * NW + wbase + l16) * DIM;

    v8f acc = {0.f, 0.f, 0.f, 0.f, 0.f, 0.f, 0.f, 0.f};
    for (int k0 = 0; k0 < DIM; k0 += 32) {
        const int ksA = k0 + hl * 8;
        const int ksB = k0 + hl * 8;
        const float4* pa  = reinterpret_cast<const float4*>(arow + ksA);
        const float4* pa2 = reinterpret_cast<const float4*>(arow + ksA + 16);
        const float4* pb  = reinterpret_cast<const float4*>(brow + ksB);
        const float4* pb2 = reinterpret_cast<const float4*>(brow + ksB + 16);
        if (k0 + 32 < DIM) __builtin_prefetch(brow + ksB + 32, 0, 3);
        v16h a, b;
        cvt8(a, 0, pa[0],  pa[1]);
        cvt8(a, 8, pa2[0], pa2[1]);
        cvt8(b, 0, pb[0],  pb[1]);
        cvt8(b, 8, pb2[0], pb2[1]);
        acc = __builtin_amdgcn_wmma_f32_16x16x32_f16(
                  false, a, false, b, (short)0, acc, false, false);
    }

    #pragma unroll
    for (int e = 0; e < 8; ++e) {
        const float d = acc[e];
        const float c = 1.0f - (d > 0.0f ? d : 0.0f);
        const int row = e + hl * 8;
        costL[row][wbase + l16] = c;
        Km[row][wbase + l16]    = __expf(c * NEG_INV_EPS);
    }

    sinkhorn_and_emd(costL, Km, uL, vL, aw, bw, red, tid,
                     k_part[i], s_l[j], &emd[i * BSZ + j]);
}

// ---------------------------------------------------------------------------
// Final loss: single block, 128 threads -> 2 scalars.
// ---------------------------------------------------------------------------
__global__ __launch_bounds__(128) void emd_loss_kernel(
    const float* __restrict__ emd, float* __restrict__ out)
{
    __shared__ float diag[BSZ];
    __shared__ float r_cs[BSZ], r_ci[BSZ], r_ss[BSZ], r_si[BSZ];
    const int t = threadIdx.x;
    diag[t] = emd[t * BSZ + t];
    __syncthreads();

    const float di = diag[t];
    float mrow = 0.f, msrow = 0.f;
    float mcol = 0.f, mscol = 0.f;
    for (int k = 0; k < BSZ; ++k) {
        if (k != t) {
            const float e  = emd[t * BSZ + k];
            const float cs = fmaxf(e - di, 0.f);
            mrow  = fmaxf(mrow, cs);
            msrow = fmaxf(msrow, (e > di) ? 0.f : cs);
            const float e2 = emd[k * BSZ + t];
            const float ci = fmaxf(e2 - di, 0.f);
            mcol  = fmaxf(mcol, ci);
            mscol = fmaxf(mscol, (e2 > di) ? 0.f : ci);
        }
    }
    r_cs[t] = mrow; r_ci[t] = mcol; r_ss[t] = msrow; r_si[t] = mscol;
    __syncthreads();
    if (t == 0) {
        float scs = 0.f, sci = 0.f, sss = 0.f, ssi = 0.f;
        for (int k = 0; k < BSZ; ++k) {
            scs += r_cs[k]; sci += r_ci[k]; sss += r_ss[k]; ssi += r_si[k];
        }
        const float inv = 1.0f / (float)BSZ;
        out[0] = sss * inv + ssi * inv;
        out[1] = scs * inv + sci * inv;
    }
}

extern "C" void kernel_launch(void* const* d_in, const int* in_sizes, int n_in,
                              void* d_out, int out_size, void* d_ws, size_t ws_size,
                              hipStream_t stream) {
    const float* im     = (const float*)d_in[0];   // [128,16,1024] f32
    const float* s      = (const float*)d_in[1];   // [128,48,1024] f32
    const int*   s_l    = (const int*)d_in[2];     // [128] i32
    const int*   k_part = (const int*)d_in[3];     // [128] i32

    const size_t n_im = (size_t)BSZ * NR * DIM;    // 2M
    const size_t n_s  = (size_t)BSZ * NW * DIM;    // 6M
    const size_t need = 65536 + (n_im + n_s) * sizeof(_Float16);  // ~16.06 MB

    float* emd = (float*)d_ws;                     // 64 KB at offset 0
    dim3 grid(BSZ, BSZ);

    if (ws_size >= need) {
        _Float16* im_h = (_Float16*)((char*)d_ws + 65536);
        _Float16* s_h  = im_h + n_im;
        const int n8_im = (int)(n_im / 8), n8_s = (int)(n_s / 8);
        cvt_f16_kernel<<<(n8_im + 255) / 256, 256, 0, stream>>>(im, im_h, n8_im);
        cvt_f16_kernel<<<(n8_s  + 255) / 256, 256, 0, stream>>>(s,  s_h,  n8_s);
        emd_pair_f16_kernel<<<grid, 96, 0, stream>>>(im_h, s_h, s_l, k_part, emd);
    } else {
        emd_pair_f32_kernel<<<grid, 96, 0, stream>>>(im, s, s_l, k_part, emd);
    }
    emd_loss_kernel<<<1, 128, 0, stream>>>(emd, (float*)d_out);
}
